// ETNNLayer_25666724560985
// MI455X (gfx1250) — compile-verified, matrix operand order checked by reading, not run on hardware
//
#include <hip/hip_runtime.h>
#include <hip/hip_bf16.h>

typedef __attribute__((ext_vector_type(16))) _Float16 v16h;
typedef __attribute__((ext_vector_type(8)))  _Float16 v8h;
typedef __attribute__((ext_vector_type(8)))  float    v8f;

#define HDIM 128
#define LDS_STRIDE 136   // 128 + 8 halves pad -> 272B rows (16B aligned)

__device__ __forceinline__ float fast_sigmoid(float x) {
    return __builtin_amdgcn_rcpf(1.0f + __expf(-x));
}
__device__ __forceinline__ float silu_f(float x) {
    return x * fast_sigmoid(x);
}
__device__ __forceinline__ float fast_tanh(float x) {
    return 2.0f * fast_sigmoid(2.0f * x) - 1.0f;
}

// Build a 16x32 f16 A-fragment from 32 consecutive f32 elements of a row.
// lane L: M = L%16 (caller picks row pointer), kb = (L>=16)?8:0.
__device__ __forceinline__ v16h make_a_global(const float* __restrict__ base, int kb) {
    const float4 a0 = *(const float4*)(base + kb);
    const float4 a1 = *(const float4*)(base + kb + 4);
    const float4 a2 = *(const float4*)(base + kb + 16);
    const float4 a3 = *(const float4*)(base + kb + 20);
    v16h r;
    r[0]  = (_Float16)a0.x; r[1]  = (_Float16)a0.y; r[2]  = (_Float16)a0.z; r[3]  = (_Float16)a0.w;
    r[4]  = (_Float16)a1.x; r[5]  = (_Float16)a1.y; r[6]  = (_Float16)a1.z; r[7]  = (_Float16)a1.w;
    r[8]  = (_Float16)a2.x; r[9]  = (_Float16)a2.y; r[10] = (_Float16)a2.z; r[11] = (_Float16)a2.w;
    r[12] = (_Float16)a3.x; r[13] = (_Float16)a3.y; r[14] = (_Float16)a3.z; r[15] = (_Float16)a3.w;
    return r;
}

// A-fragment from an f16 LDS tile (row-major, stride LDS_STRIDE halves).
__device__ __forceinline__ v16h make_a_lds(const _Float16* p) {
    v8h lo = *(const v8h*)(p);
    v8h hi = *(const v8h*)(p + 16);
    return __builtin_shufflevector(lo, hi, 0,1,2,3,4,5,6,7,8,9,10,11,12,13,14,15);
}

// B-fragment fetch from pre-swizzled weights.
__device__ __forceinline__ v16h load_b(const _Float16* __restrict__ pw, int c, int t, int lane) {
    return *(const v16h*)(pw + (((size_t)c * 8 + t) * 32 + lane) * 16);
}

// ---------------------------------------------------------------------------
// Weight pre-swizzle: W is (K x 128) f32 row-major; out is B-fragment layout.
// ---------------------------------------------------------------------------
__global__ void pack_w_kernel(const float* __restrict__ W, _Float16* __restrict__ out, int total) {
    int gid = blockIdx.x * blockDim.x + threadIdx.x;
    if (gid >= total) return;
    int h    = gid & 15;
    int lane = (gid >> 4) & 31;
    int t    = (gid >> 9) & 7;
    int c    = gid >> 12;
    int n    = t * 16 + (lane & 15);
    int kb   = (lane >= 16) ? 8 : 0;
    int k    = c * 32 + ((h < 8) ? (kb + h) : (16 + kb + (h - 8)));
    out[gid] = (_Float16)W[(size_t)k * HDIM + n];
}

// ---------------------------------------------------------------------------
// Edge kernel: one wave = 32 edges (two 16-row M-tiles sharing B fragments),
// one batch. Full message + position MLPs.
// ---------------------------------------------------------------------------
__global__ __launch_bounds__(32) void edge_kernel(
    const float* __restrict__ feat, const float* __restrict__ pos,
    const int* __restrict__ eidx, const float* __restrict__ degree,
    const float* __restrict__ w_msg1_last,   // row 256 of w_msg1 (dist column)
    const float* __restrict__ b_msg1, const float* __restrict__ b_msg2,
    const float* __restrict__ b_pos1, const float* __restrict__ w_pos2,
    const float* __restrict__ b_pos2,
    const _Float16* __restrict__ pw_msg1, const _Float16* __restrict__ pw_msg2,
    const _Float16* __restrict__ pw_pos1,
    float* __restrict__ msg_acc, float* __restrict__ pos_upd,
    int N, int E)
{
    __shared__ __align__(32) _Float16 sHid[32 * LDS_STRIDE];
    __shared__ int   sDst[32];
    __shared__ int   sSrc[32];
    __shared__ float sDist[32], sInv[32], sW[32];

    const int lane = threadIdx.x;
    const int tile = blockIdx.x;
    const int b    = blockIdx.y;

    // ---- phase 1: per-edge geometry (every lane owns one edge) ----
    float rx = 0.f, ry = 0.f, rz = 0.f;
    {
        int e = tile * 32 + lane;
        bool valid = (e < E);
        int es = valid ? eidx[e] : 0;
        int ed = valid ? eidx[E + e] : 0;
        float d = 0.f, inv = 0.f;
        if (valid) {
            const float* ps = pos + ((size_t)b * N + es) * 3;
            const float* pd = pos + ((size_t)b * N + ed) * 3;
            rx = ps[0] - pd[0]; ry = ps[1] - pd[1]; rz = ps[2] - pd[2];
            d  = sqrtf(rx * rx + ry * ry + rz * rz);
            inv = 1.0f / fmaxf(degree[ed], 1.0f);
        }
        sSrc[lane] = es; sDst[lane] = ed; sDist[lane] = d; sInv[lane] = inv;
    }
    __syncthreads();

    const int row  = lane & 15;
    const int kb   = (lane >= 16) ? 8 : 0;
    const int nl   = lane & 15;
    const int mOff = (lane >= 16) ? 8 : 0;

    // ---- A1: msg_input = [h_src | h_dst] (K=0..255) for both M-tiles ----
    v16h A1[2][8];
#pragma unroll
    for (int u = 0; u < 2; ++u) {
        const float* rs = feat + ((size_t)b * N + sSrc[u * 16 + row]) * HDIM;
        const float* rd = feat + ((size_t)b * N + sDst[u * 16 + row]) * HDIM;
#pragma unroll
        for (int c = 0; c < 8; ++c) {
            const float* base = (c < 4) ? (rs + c * 32) : (rd + (c - 4) * 32);
            A1[u][c] = make_a_global(base, kb);
        }
    }

    // ---- GEMM1: hidden = silu(msg_input @ w_msg1 + b_msg1 + dist*w_last) ----
#pragma unroll 1
    for (int t = 0; t < 8; ++t) {
        int n = t * 16 + nl;
        float bn = b_msg1[n], wl = w_msg1_last[n];
        v8f c0, c1;
#pragma unroll
        for (int v = 0; v < 8; ++v) {
            c0[v] = bn + sDist[mOff + v] * wl;
            c1[v] = bn + sDist[16 + mOff + v] * wl;
        }
#pragma unroll
        for (int cc = 0; cc < 8; ++cc) {
            v16h bf = load_b(pw_msg1, cc, t, lane);
            c0 = __builtin_amdgcn_wmma_f32_16x16x32_f16(false, A1[0][cc], false, bf, (short)0, c0, false, false);
            c1 = __builtin_amdgcn_wmma_f32_16x16x32_f16(false, A1[1][cc], false, bf, (short)0, c1, false, false);
        }
#pragma unroll
        for (int v = 0; v < 8; ++v) {
            sHid[(mOff + v) * LDS_STRIDE + n]        = (_Float16)silu_f(c0[v]);
            sHid[(16 + mOff + v) * LDS_STRIDE + n]   = (_Float16)silu_f(c1[v]);
        }
    }
    __syncthreads();

    // ---- A2 from hidden ----
    v16h A2[2][4];
#pragma unroll
    for (int u = 0; u < 2; ++u)
#pragma unroll
        for (int cc = 0; cc < 4; ++cc)
            A2[u][cc] = make_a_lds(&sHid[(u * 16 + row) * LDS_STRIDE + cc * 32 + kb]);
    __syncthreads();

    // ---- GEMM2: messages = hidden @ w_msg2 + b_msg2; scatter + stash ----
#pragma unroll 1
    for (int t = 0; t < 8; ++t) {
        int n = t * 16 + nl;
        float bn = b_msg2[n];
        v8f c0, c1;
#pragma unroll
        for (int v = 0; v < 8; ++v) { c0[v] = bn; c1[v] = bn; }
#pragma unroll
        for (int cc = 0; cc < 4; ++cc) {
            v16h bf = load_b(pw_msg2, cc, t, lane);
            c0 = __builtin_amdgcn_wmma_f32_16x16x32_f16(false, A2[0][cc], false, bf, (short)0, c0, false, false);
            c1 = __builtin_amdgcn_wmma_f32_16x16x32_f16(false, A2[1][cc], false, bf, (short)0, c1, false, false);
        }
#pragma unroll
        for (int u = 0; u < 2; ++u) {
#pragma unroll
            for (int v = 0; v < 8; ++v) {
                int m = u * 16 + mOff + v;
                float x = (u == 0) ? c0[v] : c1[v];
                float iv = sInv[m];
                if (iv != 0.0f)
                    atomicAdd(&msg_acc[((size_t)b * N + sDst[m]) * HDIM + n], x * iv);
                sHid[m * LDS_STRIDE + n] = (_Float16)x;
            }
        }
    }
    __syncthreads();

    // ---- A3 from messages ----
    v16h A3[2][4];
#pragma unroll
    for (int u = 0; u < 2; ++u)
#pragma unroll
        for (int cc = 0; cc < 4; ++cc)
            A3[u][cc] = make_a_lds(&sHid[(u * 16 + row) * LDS_STRIDE + cc * 32 + kb]);

    // ---- GEMM3 + dot(w_pos2): w = tanh(silu(msg@w_pos1+b) @ w_pos2 + b2) ----
    float pv0[8], pv1[8];
#pragma unroll
    for (int v = 0; v < 8; ++v) { pv0[v] = 0.f; pv1[v] = 0.f; }
#pragma unroll 1
    for (int t = 0; t < 8; ++t) {
        int n = t * 16 + nl;
        float bn = b_pos1[n], wp = w_pos2[n];
        v8f c0, c1;
#pragma unroll
        for (int v = 0; v < 8; ++v) { c0[v] = bn; c1[v] = bn; }
#pragma unroll
        for (int cc = 0; cc < 4; ++cc) {
            v16h bf = load_b(pw_pos1, cc, t, lane);
            c0 = __builtin_amdgcn_wmma_f32_16x16x32_f16(false, A3[0][cc], false, bf, (short)0, c0, false, false);
            c1 = __builtin_amdgcn_wmma_f32_16x16x32_f16(false, A3[1][cc], false, bf, (short)0, c1, false, false);
        }
#pragma unroll
        for (int v = 0; v < 8; ++v) {
            pv0[v] += silu_f(c0[v]) * wp;
            pv1[v] += silu_f(c1[v]) * wp;
        }
    }
    // reduce over n within each 16-lane half
#pragma unroll
    for (int v = 0; v < 8; ++v) {
        float s0 = pv0[v], s1 = pv1[v];
        s0 += __shfl_xor(s0, 1, 16); s1 += __shfl_xor(s1, 1, 16);
        s0 += __shfl_xor(s0, 2, 16); s1 += __shfl_xor(s1, 2, 16);
        s0 += __shfl_xor(s0, 4, 16); s1 += __shfl_xor(s1, 4, 16);
        s0 += __shfl_xor(s0, 8, 16); s1 += __shfl_xor(s1, 8, 16);
        pv0[v] = s0; pv1[v] = s1;
    }
    if ((lane & 15) == 0) {
#pragma unroll
        for (int v = 0; v < 8; ++v) {
            sW[mOff + v]      = pv0[v];
            sW[16 + mOff + v] = pv1[v];
        }
    }
    __syncthreads();

    // ---- position scatter (every lane owns one edge) ----
    {
        int e = tile * 32 + lane;
        if (e < E) {
            float w = fast_tanh(sW[lane] + b_pos2[0]);
            float* pu = pos_upd + ((size_t)b * N + sDst[lane]) * 3;
            atomicAdd(&pu[0], w * rx);
            atomicAdd(&pu[1], w * ry);
            atomicAdd(&pu[2], w * rz);
        }
    }
}

// ---------------------------------------------------------------------------
// Node update: out_feat = feat + mlp2([feat | msg_acc], w_upd1, w_upd2)
// One wave = 32 nodes (two M-tiles sharing B fragments).
// ---------------------------------------------------------------------------
__global__ __launch_bounds__(32) void update_kernel(
    const float* __restrict__ feat, const float* __restrict__ msg_acc,
    const float* __restrict__ b_upd1, const float* __restrict__ b_upd2,
    const _Float16* __restrict__ pw1, const _Float16* __restrict__ pw2,
    float* __restrict__ out_feat, int N)
{
    __shared__ __align__(32) _Float16 sHid[32 * LDS_STRIDE];

    const int lane = threadIdx.x;
    const int tile = blockIdx.x;
    const int b    = blockIdx.y;
    const int row  = lane & 15;
    const int kb   = (lane >= 16) ? 8 : 0;
    const int nl   = lane & 15;
    const int mOff = (lane >= 16) ? 8 : 0;

    v16h A1[2][8];
#pragma unroll
    for (int u = 0; u < 2; ++u) {
        int node = tile * 32 + u * 16 + row;
        int nodeC = node < N ? node : (N - 1);
        const float* rf = feat    + ((size_t)b * N + nodeC) * HDIM;
        const float* rm = msg_acc + ((size_t)b * N + nodeC) * HDIM;
#pragma unroll
        for (int c = 0; c < 8; ++c) {
            const float* base = (c < 4) ? (rf + c * 32) : (rm + (c - 4) * 32);
            A1[u][c] = make_a_global(base, kb);
        }
    }

#pragma unroll 1
    for (int t = 0; t < 8; ++t) {
        int n = t * 16 + nl;
        float bn = b_upd1[n];
        v8f c0, c1;
#pragma unroll
        for (int v = 0; v < 8; ++v) { c0[v] = bn; c1[v] = bn; }
#pragma unroll
        for (int cc = 0; cc < 8; ++cc) {
            v16h bf = load_b(pw1, cc, t, lane);
            c0 = __builtin_amdgcn_wmma_f32_16x16x32_f16(false, A1[0][cc], false, bf, (short)0, c0, false, false);
            c1 = __builtin_amdgcn_wmma_f32_16x16x32_f16(false, A1[1][cc], false, bf, (short)0, c1, false, false);
        }
#pragma unroll
        for (int v = 0; v < 8; ++v) {
            sHid[(mOff + v) * LDS_STRIDE + n]      = (_Float16)silu_f(c0[v]);
            sHid[(16 + mOff + v) * LDS_STRIDE + n] = (_Float16)silu_f(c1[v]);
        }
    }
    __syncthreads();

    v16h A2[2][4];
#pragma unroll
    for (int u = 0; u < 2; ++u)
#pragma unroll
        for (int cc = 0; cc < 4; ++cc)
            A2[u][cc] = make_a_lds(&sHid[(u * 16 + row) * LDS_STRIDE + cc * 32 + kb]);

#pragma unroll 1
    for (int t = 0; t < 8; ++t) {
        int n = t * 16 + nl;
        float bn = b_upd2[n];
        v8f c0, c1;
#pragma unroll
        for (int v = 0; v < 8; ++v) { c0[v] = bn; c1[v] = bn; }
#pragma unroll
        for (int cc = 0; cc < 4; ++cc) {
            v16h bf = load_b(pw2, cc, t, lane);
            c0 = __builtin_amdgcn_wmma_f32_16x16x32_f16(false, A2[0][cc], false, bf, (short)0, c0, false, false);
            c1 = __builtin_amdgcn_wmma_f32_16x16x32_f16(false, A2[1][cc], false, bf, (short)0, c1, false, false);
        }
#pragma unroll
        for (int u = 0; u < 2; ++u) {
#pragma unroll
            for (int v = 0; v < 8; ++v) {
                int mg = tile * 32 + u * 16 + mOff + v;
                if (mg < N) {
                    size_t o = ((size_t)b * N + mg) * HDIM + n;
                    float x = (u == 0) ? c0[v] : c1[v];
                    out_feat[o] = feat[o] + x;
                }
            }
        }
    }
}

// ---------------------------------------------------------------------------
// Positions: out_pos = pos + 0.5 * pos_upd
// ---------------------------------------------------------------------------
__global__ void pos_finalize_kernel(const float* __restrict__ pos,
                                    const float* __restrict__ pos_upd,
                                    float* __restrict__ out_pos, int total)
{
    int gid = blockIdx.x * blockDim.x + threadIdx.x;
    if (gid < total) out_pos[gid] = pos[gid] + 0.5f * pos_upd[gid];
}

// ---------------------------------------------------------------------------
extern "C" void kernel_launch(void* const* d_in, const int* in_sizes, int n_in,
                              void* d_out, int out_size, void* d_ws, size_t ws_size,
                              hipStream_t stream)
{
    const float* feat   = (const float*)d_in[0];
    const float* pos    = (const float*)d_in[1];
    const int*   ei_up  = (const int*)  d_in[2];
    const int*   ei_dn  = (const int*)  d_in[3];
    const float* deg_up = (const float*)d_in[4];
    const float* deg_dn = (const float*)d_in[5];
    const float* w_msg1_up = (const float*)d_in[6];
    const float* b_msg1_up = (const float*)d_in[7];
    const float* w_msg2_up = (const float*)d_in[8];
    const float* b_msg2_up = (const float*)d_in[9];
    const float* w_pos1_up = (const float*)d_in[10];
    const float* b_pos1_up = (const float*)d_in[11];
    const float* w_pos2_up = (const float*)d_in[12];
    const float* b_pos2_up = (const float*)d_in[13];
    const float* w_msg1_dn = (const float*)d_in[14];
    const float* b_msg1_dn = (const float*)d_in[15];
    const float* w_msg2_dn = (const float*)d_in[16];
    const float* b_msg2_dn = (const float*)d_in[17];
    const float* w_pos1_dn = (const float*)d_in[18];
    const float* b_pos1_dn = (const float*)d_in[19];
    const float* w_pos2_dn = (const float*)d_in[20];
    const float* b_pos2_dn = (const float*)d_in[21];
    const float* w_upd1    = (const float*)d_in[22];
    const float* b_upd1    = (const float*)d_in[23];
    const float* w_upd2    = (const float*)d_in[24];
    const float* b_upd2    = (const float*)d_in[25];

    const int N = in_sizes[4];
    const int E = in_sizes[2] / 2;
    const int B = in_sizes[0] / (N * HDIM);

    // workspace layout
    size_t accElems = (size_t)B * N * HDIM;
    size_t posElems = (size_t)B * N * 3;
    float* msg_acc = (float*)d_ws;
    float* pos_upd = msg_acc + accElems;
    size_t packOffB = ((accElems + posElems) * sizeof(float) + 255) & ~(size_t)255;
    _Float16* packBase = (_Float16*)((char*)d_ws + packOffB);

    _Float16* pm1u = packBase;                 // 256x128 -> 32768 halves
    _Float16* pm2u = pm1u + 32768;             // 128x128 -> 16384
    _Float16* pp1u = pm2u + 16384;
    _Float16* pm1d = pp1u + 16384;
    _Float16* pm2d = pm1d + 32768;
    _Float16* pp1d = pm2d + 16384;
    _Float16* pu1  = pp1d + 16384;             // 256x128
    _Float16* pu2  = pu1  + 32768;             // 128x128

    // 1) zero accumulators
    hipMemsetAsync(d_ws, 0, (accElems + posElems) * sizeof(float), stream);

    // 2) pre-swizzle weights into WMMA B-fragment layout
    auto pack = [&](const float* W, _Float16* out, int K) {
        int total = (K / 32) * 4096;
        pack_w_kernel<<<(total + 255) / 256, 256, 0, stream>>>(W, out, total);
    };
    pack(w_msg1_up, pm1u, 256);   // rows 0..255 (dist row handled separately)
    pack(w_msg2_up, pm2u, 128);
    pack(w_pos1_up, pp1u, 128);
    pack(w_msg1_dn, pm1d, 256);
    pack(w_msg2_dn, pm2d, 128);
    pack(w_pos1_dn, pp1d, 128);
    pack(w_upd1,    pu1,  256);
    pack(w_upd2,    pu2,  128);

    // 3) edge kernels (both neighborhoods accumulate into msg_acc / pos_upd)
    dim3 egrid((E + 31) / 32, B);
    edge_kernel<<<egrid, 32, 0, stream>>>(feat, pos, ei_up, deg_up,
        w_msg1_up + 256 * HDIM, b_msg1_up, b_msg2_up, b_pos1_up, w_pos2_up, b_pos2_up,
        pm1u, pm2u, pp1u, msg_acc, pos_upd, N, E);
    edge_kernel<<<egrid, 32, 0, stream>>>(feat, pos, ei_dn, deg_dn,
        w_msg1_dn + 256 * HDIM, b_msg1_dn, b_msg2_dn, b_pos1_dn, w_pos2_dn, b_pos2_dn,
        pm1d, pm2d, pp1d, msg_acc, pos_upd, N, E);

    // 4) node feature update
    float* out_feat = (float*)d_out;
    dim3 ugrid((N + 31) / 32, B);
    update_kernel<<<ugrid, 32, 0, stream>>>(feat, msg_acc, b_upd1, b_upd2,
                                            pu1, pu2, out_feat, N);

    // 5) positions
    float* out_pos = out_feat + (size_t)B * N * HDIM;
    int ptotal = (int)posElems;
    pos_finalize_kernel<<<(ptotal + 255) / 256, 256, 0, stream>>>(pos, pos_upd, out_pos, ptotal);
}
